// AGAOperator_50912542326914
// MI455X (gfx1250) — matrix operand order repaired and checked by reading, not compiled, not run on hardware
//
#include <hip/hip_runtime.h>
#include <math.h>

// ---------------------------------------------------------------------------
// B=4, S=2048, H=2048, Db=128, Dv=512, K_top=8.  T = B*S = 8192 tokens.
//   (1) Q = HS @ Wq^T              : bf16 WMMA GEMM, M=8192 N=128  K=2048
//   (2) gather/softmax/weighted sum: bandwidth kernel (~536MB gathers)
//   (3) mid = gelu(aux @ Wdown^T)  : bf16 WMMA GEMM, M=8192 N=512  K=2048
//   (4) out = primary+g*(mid@WupT) : bf16 WMMA GEMM, M=8192 N=2048 K=512
// Roofline: ~39 GFLOP bf16-WMMA vs ~740MB HBM traffic -> both ~30us.
// Single-buffer K-loop: backend software-pipelines it itself (verified in
// round-1 asm) and stays ~110 VGPRs -> high occupancy, no vgpr-msb traffic,
// no WMMA->VALU hazard nops.
// ---------------------------------------------------------------------------

typedef __bf16 bf16_t;
typedef bf16_t v16bf __attribute__((ext_vector_type(16)));
typedef bf16_t v8bf  __attribute__((ext_vector_type(8)));
typedef bf16_t v4bf  __attribute__((ext_vector_type(4)));
typedef float  v8f   __attribute__((ext_vector_type(8)));

// ---------------------------------------------------------------------------
// fp32 -> bf16 elementwise convert (vectorized x4)
// ---------------------------------------------------------------------------
__global__ __launch_bounds__(256)
void cvt_f32_bf16(const float* __restrict__ in, bf16_t* __restrict__ out, int n4) {
  int i = blockIdx.x * blockDim.x + threadIdx.x;
  if (i < n4) {
    float4 v = ((const float4*)in)[i];
    v4bf o;
    o[0] = (bf16_t)v.x; o[1] = (bf16_t)v.y; o[2] = (bf16_t)v.z; o[3] = (bf16_t)v.w;
    ((v4bf*)out)[i] = o;
  }
}

// ---------------------------------------------------------------------------
// fp32 [R,C] row-major -> bf16 [C,R] row-major (weights, small)
// ---------------------------------------------------------------------------
__global__ __launch_bounds__(256)
void transpose_to_bf16(const float* __restrict__ in, bf16_t* __restrict__ out,
                       int R, int C) {
  int idx = blockIdx.x * blockDim.x + threadIdx.x;
  if (idx < R * C) {
    int r = idx / C;
    int c = idx - r * C;
    out[(size_t)c * R + r] = (bf16_t)in[idx];
  }
}

// ---------------------------------------------------------------------------
// GEMM: out[m,n] = sum_k A[m,k]*Bt[k,n];  A bf16 [M,K], Bt bf16 [K,N].
// Block = 256 threads = 8 waves laid out 2(M) x 4(N); wave tile 64x32;
// block tile 128x128.  K multiple of 32.
// Fragment layouts (CDNA5 wave32):
//  A 16x32:  lane<16 -> row M=lane,    K in {0..7, 16..23}
//            lane>=16-> row M=lane-16, K in {8..15, 24..31}
//  B 32x16:  lane L -> N col L%16, K in same set as A (from Bt [K,N])
// Epilogue by MODE:
//   MODE 0: outF fp32
//   MODE 1: outB = bf16(gelu_exact(acc))
//   MODE 2: outF = primary + gate[m]*acc
// ---------------------------------------------------------------------------
template <int MODE>
__global__ __launch_bounds__(256)
void gemm_bf16_wmma(const bf16_t* __restrict__ A, const bf16_t* __restrict__ Bt,
                    int M, int N, int K,
                    float* __restrict__ outF, bf16_t* __restrict__ outB,
                    const float* __restrict__ primary, const float* __restrict__ gate) {
  const int lane  = threadIdx.x & 31;
  const int wave  = threadIdx.x >> 5;
  const int mBase = blockIdx.y * 128 + (wave >> 2) * 64;
  const int nBase = blockIdx.x * 128 + (wave & 3) * 32;
  const int half  = lane >> 4;
  const int mrow  = lane & 15;

  v8f acc[4][2];
#pragma unroll
  for (int i = 0; i < 4; ++i)
#pragma unroll
    for (int j = 0; j < 2; ++j) acc[i][j] = (v8f)0.0f;

  int aOff[4];
#pragma unroll
  for (int i = 0; i < 4; ++i)
    aOff[i] = (mBase + i * 16 + mrow) * K + half * 8;
  const int bOff = half * 8 * N + nBase + mrow;

  for (int k = 0; k < K; k += 32) {
    // speculative GL2 prefetch two K-steps ahead (unconditional: speculative
    // prefetch silently drops failed translations, safe past buffer end)
    __builtin_prefetch(A + aOff[0] + k + 64, 0, 0);
    __builtin_prefetch(Bt + bOff + (k + 64) * N, 0, 0);

    v16bf a[4], b[2];
#pragma unroll
    for (int i = 0; i < 4; ++i) {
      union { v16bf v; v8bf h[2]; } u;
      u.h[0] = *(const v8bf*)(A + aOff[i] + k);
      u.h[1] = *(const v8bf*)(A + aOff[i] + k + 16);
      a[i] = u.v;
    }
#pragma unroll
    for (int j = 0; j < 2; ++j)
#pragma unroll
      for (int s = 0; s < 16; ++s)
        b[j][s] = Bt[bOff + (k + (s < 8 ? s : s + 8)) * N + j * 16];

#pragma unroll
    for (int i = 0; i < 4; ++i)
#pragma unroll
      for (int j = 0; j < 2; ++j)
        acc[i][j] = __builtin_amdgcn_wmma_f32_16x16x32_bf16(
            false, a[i], false, b[j], (short)0, acc[i][j], false, false);
  }

  // C/D layout: VGPR r -> (M = r + half*8, N = lane&15) within the 16x16 tile
#pragma unroll
  for (int i = 0; i < 4; ++i) {
#pragma unroll
    for (int r = 0; r < 8; ++r) {
      const int m = mBase + i * 16 + half * 8 + r;
#pragma unroll
      for (int j = 0; j < 2; ++j) {
        const int n = nBase + j * 16 + mrow;
        const float v = acc[i][j][r];
        const int o = m * N + n;
        if (MODE == 0) {
          outF[o] = v;
        } else if (MODE == 1) {
          const float g = 0.5f * v * (1.0f + erff(v * 0.70710678118654752f));
          outB[o] = (bf16_t)g;
        } else {
          outF[o] = primary[o] + gate[m] * v;
        }
      }
    }
  }
}

// ---------------------------------------------------------------------------
// Gather + scores + softmax + weighted value sum.  One block (256 thr) per
// token; wave w (of 8) computes score w via a wave32 shuffle reduction; then
// all threads do the float4-vectorized weighted sum over the 8 gathered
// 2048-wide value rows.  aux output in bf16 (feeds GEMM 3).
// ---------------------------------------------------------------------------
__global__ __launch_bounds__(256)
void gather_attend(const float* __restrict__ Q,        // [T,128]
                   const int* __restrict__ topIdx,     // [T,8]
                   const float* __restrict__ slotK,    // [Ns,128]
                   const float* __restrict__ slotV,    // [Ns,2048]
                   bf16_t* __restrict__ aux,           // [T,2048]
                   int H) {
  const int t    = blockIdx.x;
  const int lane = threadIdx.x & 31;
  const int wave = threadIdx.x >> 5;

  __shared__ float s_scores[8];
  __shared__ int   s_idx[8];
  if (threadIdx.x < 8) s_idx[threadIdx.x] = topIdx[t * 8 + threadIdx.x];
  __syncthreads();

  {
    const int idx = s_idx[wave];
    const float* qp = Q + (size_t)t * 128;
    const float* kp = slotK + (size_t)idx * 128;
    float p = 0.0f;
#pragma unroll
    for (int i = 0; i < 4; ++i) {
      const int c = lane + i * 32;
      p += qp[c] * kp[c];
    }
#pragma unroll
    for (int off = 16; off > 0; off >>= 1) p += __shfl_xor(p, off, 32);
    if (lane == 0) s_scores[wave] = p * 0.08838834764831845f;  // 1/sqrt(128)
  }
  __syncthreads();

  float w[8];
  int   idx[8];
  {
    float mx = -3.4e38f;
#pragma unroll
    for (int kk = 0; kk < 8; ++kk) { idx[kk] = s_idx[kk]; mx = fmaxf(mx, s_scores[kk]); }
    float sum = 0.0f;
#pragma unroll
    for (int kk = 0; kk < 8; ++kk) { w[kk] = __expf(s_scores[kk] - mx); sum += w[kk]; }
    const float inv = 1.0f / sum;
#pragma unroll
    for (int kk = 0; kk < 8; ++kk) w[kk] *= inv;
  }

  // float4-vectorized weighted sum: 2 passes over H=2048 with 256 threads
  for (int h = threadIdx.x * 4; h < H; h += 256 * 4) {
    float ax = 0.0f, ay = 0.0f, az = 0.0f, aw = 0.0f;
#pragma unroll
    for (int kk = 0; kk < 8; ++kk) {
      const float4 v = *(const float4*)(slotV + (size_t)idx[kk] * H + h);
      ax += w[kk] * v.x; ay += w[kk] * v.y; az += w[kk] * v.z; aw += w[kk] * v.w;
    }
    v4bf o;
    o[0] = (bf16_t)ax; o[1] = (bf16_t)ay; o[2] = (bf16_t)az; o[3] = (bf16_t)aw;
    *(v4bf*)(aux + (size_t)t * H + h) = o;
  }
}

// ---------------------------------------------------------------------------
// Host-side launcher
// ---------------------------------------------------------------------------
extern "C" void kernel_launch(void* const* d_in, const int* in_sizes, int n_in,
                              void* d_out, int out_size, void* d_ws, size_t ws_size,
                              hipStream_t stream) {
  const float* hs      = (const float*)d_in[0];  // [4,2048,2048]
  const float* primary = (const float*)d_in[1];  // [4,2048,2048]
  const float* gate    = (const float*)d_in[2];  // [4,2048]
  const int*   topIdx  = (const int*)  d_in[3];  // [4,2048,8]
  const float* slotK   = (const float*)d_in[4];  // [50000,128]
  const float* slotV   = (const float*)d_in[5];  // [50000,2048]
  const float* Wq      = (const float*)d_in[6];  // [128,2048]
  const float* Wdown   = (const float*)d_in[7];  // [512,2048]
  const float* Wup     = (const float*)d_in[8];  // [2048,512]
  float* out = (float*)d_out;

  const int T = 4 * 2048;   // tokens
  const int H = 2048, Db = 128, Dv = 512;

  // workspace carve-up (all 256B-aligned)
  char* ws = (char*)d_ws;
  bf16_t* hs_bf   = (bf16_t*)(ws);                         // 32 MB   [T,H]
  bf16_t* WqT     = (bf16_t*)(ws + 33554432);              // 512 KB  [H,Db]
  bf16_t* WdownT  = (bf16_t*)(ws + 34078720);              // 2 MB    [H,Dv]
  bf16_t* WupT    = (bf16_t*)(ws + 36175872);              // 2 MB    [Dv,H]
  float*  Qbuf    = (float*) (ws + 38273024);              // 4 MB    [T,Db]
  bf16_t* aux1    = (bf16_t*)(ws + 42467328);              // 32 MB   [T,H]
  bf16_t* mid     = (bf16_t*)(ws + 76021760);              // 8 MB    [T,Dv]

  // (0) one-time precision conversions
  {
    const int n4 = (T * H) / 4;
    cvt_f32_bf16<<<(n4 + 255) / 256, 256, 0, stream>>>(hs, hs_bf, n4);
    transpose_to_bf16<<<(Db * H + 255) / 256, 256, 0, stream>>>(Wq, WqT, Db, H);
    transpose_to_bf16<<<(Dv * H + 255) / 256, 256, 0, stream>>>(Wdown, WdownT, Dv, H);
    transpose_to_bf16<<<(H * Dv + 255) / 256, 256, 0, stream>>>(Wup, WupT, H, Dv);
  }

  // (1) Q = HS @ Wq^T   (M=8192, N=128, K=2048)
  gemm_bf16_wmma<0><<<dim3(Db / 128, T / 128), 256, 0, stream>>>(
      hs_bf, WqT, T, Db, H, Qbuf, nullptr, nullptr, nullptr);

  // (2) gather + softmax + weighted sum -> aux1 (bf16)
  gather_attend<<<T, 256, 0, stream>>>(Qbuf, topIdx, slotK, slotV, aux1, H);

  // (3) mid = gelu(aux1 @ Wdown^T)   (M=8192, N=512, K=2048)
  gemm_bf16_wmma<1><<<dim3(Dv / 128, T / 128), 256, 0, stream>>>(
      aux1, WdownT, T, Dv, H, nullptr, mid, nullptr, nullptr);

  // (4) out = primary + gate * (mid @ Wup^T)   (M=8192, N=2048, K=512)
  gemm_bf16_wmma<2><<<dim3(H / 128, T / 128), 256, 0, stream>>>(
      mid, WupT, T, H, Dv, out, nullptr, primary, gate);
}
